// VL_SAE_62010737819896
// MI455X (gfx1250) — matrix operand order repaired and checked by reading, not compiled, not run on hardware
//
#include <hip/hip_runtime.h>
#include <hip/hip_bf16.h>

typedef __attribute__((ext_vector_type(16))) _Float16 v16h;
typedef __attribute__((ext_vector_type(8)))  _Float16 v8h;
typedef __attribute__((ext_vector_type(8)))  float    v8f;

#define B_DIM 4096
#define D_DIM 1024
#define H_DIM 16384
#define TOPK  32
#define LDB   40   // LDS row stride in halves (80B, multiple of 16B, conflict-free)

// ---------------------------------------------------------------------------
// 1. Normalize embeddings -> f16 [2*B, D] row-major (A-matrix source)
// ---------------------------------------------------------------------------
__global__ __launch_bounds__(256) void prep_emb(const float* __restrict__ v,
                                                const float* __restrict__ t,
                                                _Float16* __restrict__ ef16) {
  int row = blockIdx.x;
  int m   = blockIdx.y;
  const float* src = (m ? t : v) + (size_t)row * D_DIM;
  _Float16* dst = ef16 + ((size_t)m * B_DIM + row) * D_DIM;
  __shared__ float red[256];
  float ss = 0.0f;
  for (int i = threadIdx.x; i < D_DIM; i += 256) { float x = src[i]; ss += x * x; }
  red[threadIdx.x] = ss;
  __syncthreads();
  for (int s = 128; s > 0; s >>= 1) {
    if (threadIdx.x < s) red[threadIdx.x] += red[threadIdx.x + s];
    __syncthreads();
  }
  float scale = 1.0f / fmaxf(sqrtf(red[0]), 1e-12f);
  for (int i = threadIdx.x; i < D_DIM; i += 256)
    dst[i] = (_Float16)(src[i] * scale);
}

// ---------------------------------------------------------------------------
// 2. Normalize encoder rows -> f16 in pre-swizzled B layout:
//    wTs[((d>>5)*H + h)*32 + (d&31)]  (32 contiguous k-halves per column)
//    so GEMM staging is pure b128 loads/stores.
// ---------------------------------------------------------------------------
__global__ __launch_bounds__(256) void prep_encoder(const float* __restrict__ w,
                                                    _Float16* __restrict__ wTs) {
  int h0  = blockIdx.x * 32;
  int tid = threadIdx.x;
  int lane = tid & 31, wv = tid >> 5;           // 8 waves, 4 rows each
  __shared__ float inv[32];
  __shared__ float tile[32][33];

  for (int rr = 0; rr < 4; ++rr) {
    int h = h0 + wv * 4 + rr;
    const float* src = w + (size_t)h * D_DIM;
    float ss = 0.0f;
    for (int i = lane; i < D_DIM; i += 32) { float x = src[i]; ss += x * x; }
    #pragma unroll
    for (int msk = 16; msk; msk >>= 1) ss += __shfl_xor(ss, msk, 32);
    if (lane == 0) inv[wv * 4 + rr] = 1.0f / fmaxf(sqrtf(ss), 1e-12f);
  }
  __syncthreads();

  for (int d0 = 0; d0 < D_DIM; d0 += 32) {
    #pragma unroll
    for (int p = 0; p < 4; ++p) {               // coalesced read: 32 d contiguous
      int hi = p * 8 + (tid >> 5), dj = tid & 31;
      tile[dj][hi] = w[(size_t)(h0 + hi) * D_DIM + d0 + dj];
    }
    __syncthreads();
    #pragma unroll
    for (int p = 0; p < 4; ++p) {               // coalesced write: 32 k contiguous
      int hi = p * 8 + (tid >> 5), dj = tid & 31;
      wTs[((size_t)(d0 >> 5) * H_DIM + h0 + hi) * 32 + dj] =
          (_Float16)(tile[dj][hi] * inv[hi]);
    }
    __syncthreads();
  }
}

// ---------------------------------------------------------------------------
// 3. WMMA GEMM + fused activation: act[b,h] = 2 - sqrt(2 - 2*clip(cos))
//    128(M) x 256(N) tile per WG, 8 waves, wave = 32x128 (2x8 frags), K=32
// ---------------------------------------------------------------------------
__global__ __launch_bounds__(256) void gemm_act(const _Float16* __restrict__ ef16,
                                                const _Float16* __restrict__ wTs,
                                                float* __restrict__ latent) {
  int h0 = blockIdx.x * 256;
  int b0 = blockIdx.y * 128;
  int m  = blockIdx.z;
  const _Float16* A = ef16 + (size_t)m * B_DIM * D_DIM;
  float* out = latent + (size_t)m * B_DIM * H_DIM;

  __shared__ __align__(16) _Float16 As[2][128][LDB];  // [row(b)][k]
  __shared__ __align__(16) _Float16 Bs[2][256][LDB];  // [col(h)][k]

  int tid  = threadIdx.x;
  int lane = tid & 31;
  int wave = tid >> 5;
  int wm = wave & 3, wn = wave >> 2;
  int l16 = lane & 15, half = lane >> 4;

  v8f zero = {};
  v8f acc[2][8];
  #pragma unroll
  for (int i = 0; i < 2; ++i)
    #pragma unroll
    for (int j = 0; j < 8; ++j) acc[i][j] = zero;

  // stage tile 0
  #pragma unroll
  for (int i = 0; i < 2; ++i) {
    int g = tid + i * 256;
    int r = g >> 2, part = g & 3;
    *(v8h*)&As[0][r][part * 8] =
        *(const v8h*)(A + (size_t)(b0 + r) * D_DIM + part * 8);
  }
  #pragma unroll
  for (int i = 0; i < 4; ++i) {
    int g = tid + i * 256;
    int col = g >> 2, part = g & 3;
    *(v8h*)&Bs[0][col][part * 8] =
        *(const v8h*)(wTs + ((size_t)0 * H_DIM + h0 + col) * 32 + part * 8);
  }
  __syncthreads();

  int buf = 0;
  v8h pa[2], pb[4];
  const int NT = D_DIM / 32;
  for (int kt = 0; kt < NT; ++kt) {
    if (kt + 1 < NT) {                           // prefetch next tile into regs
      int kn = (kt + 1) * 32;
      #pragma unroll
      for (int i = 0; i < 2; ++i) {
        int g = tid + i * 256;
        int r = g >> 2, part = g & 3;
        pa[i] = *(const v8h*)(A + (size_t)(b0 + r) * D_DIM + kn + part * 8);
      }
      #pragma unroll
      for (int i = 0; i < 4; ++i) {
        int g = tid + i * 256;
        int col = g >> 2, part = g & 3;
        pb[i] = *(const v8h*)(wTs + ((size_t)(kt + 1) * H_DIM + h0 + col) * 32 + part * 8);
      }
    }
    if (kt + 2 < NT)                             // L2 prefetch of tile kt+2
      __builtin_prefetch(wTs + ((size_t)(kt + 2) * H_DIM + h0) * 32 + tid * 32, 0, 1);

    // build fragments per ISA wave32 layouts
    v16h af[2], bf[8];
    #pragma unroll
    for (int mi = 0; mi < 2; ++mi) {
      int row = wm * 32 + mi * 16 + l16;
      v8h lo = *(const v8h*)&As[buf][row][half * 8];       // K 0..7  / 8..15
      v8h hi = *(const v8h*)&As[buf][row][16 + half * 8];  // K 16..23 / 24..31
      #pragma unroll
      for (int j = 0; j < 8; ++j) { af[mi][j] = lo[j]; af[mi][j + 8] = hi[j]; }
    }
    #pragma unroll
    for (int ni = 0; ni < 8; ++ni) {
      int col = wn * 128 + ni * 16 + l16;
      v8h lo = *(const v8h*)&Bs[buf][col][half * 8];       // K 0..7  / 8..15
      v8h hi = *(const v8h*)&Bs[buf][col][16 + half * 8];  // K 16..23 / 24..31
      #pragma unroll
      for (int j = 0; j < 8; ++j) { bf[ni][j] = lo[j]; bf[ni][j + 8] = hi[j]; }
    }

    #pragma unroll
    for (int mi = 0; mi < 2; ++mi)
      #pragma unroll
      for (int ni = 0; ni < 8; ++ni)
        acc[mi][ni] = __builtin_amdgcn_wmma_f32_16x16x32_f16(
            false, af[mi], false, bf[ni], (short)0, acc[mi][ni], false, false);

    if (kt + 1 < NT) {
      __syncthreads();
      #pragma unroll
      for (int i = 0; i < 2; ++i) {
        int g = tid + i * 256;
        int r = g >> 2, part = g & 3;
        *(v8h*)&As[buf ^ 1][r][part * 8] = pa[i];
      }
      #pragma unroll
      for (int i = 0; i < 4; ++i) {
        int g = tid + i * 256;
        int col = g >> 2, part = g & 3;
        *(v8h*)&Bs[buf ^ 1][col][part * 8] = pb[i];
      }
      __syncthreads();
      buf ^= 1;
    }
  }

  // epilogue: C layout VGPR r -> (M = r + 8*half, N = l16); fused activation
  #pragma unroll
  for (int mi = 0; mi < 2; ++mi)
    #pragma unroll
    for (int ni = 0; ni < 8; ++ni) {
      int col = h0 + wn * 128 + ni * 16 + l16;
      #pragma unroll
      for (int r = 0; r < 8; ++r) {
        int row = b0 + wm * 32 + mi * 16 + half * 8 + r;
        float c = acc[mi][ni][r];
        c = fminf(fmaxf(c, -1.0f), 1.0f);
        float a = 2.0f - sqrtf(fmaxf(2.0f - 2.0f * c, 0.0f));
        out[(size_t)row * H_DIM + col] = a;
      }
    }
}

// ---------------------------------------------------------------------------
// 4. Per-row top-32 via 12-bit histogram select (act in [0,2], monotone key)
// ---------------------------------------------------------------------------
__global__ __launch_bounds__(256) void topk_select(float* __restrict__ latent,
                                                   int* __restrict__ idx_list,
                                                   float* __restrict__ val_list) {
  int row = blockIdx.x;                                   // 0..2B-1
  float* p = latent + (size_t)row * H_DIM;
  __shared__ int hist[4096];
  __shared__ int s_thr, s_above, s_eq, s_slot;
  for (int i = threadIdx.x; i < 4096; i += 256) hist[i] = 0;
  if (threadIdx.x == 0) { s_eq = 0; s_slot = 0; }
  __syncthreads();

  for (int i = threadIdx.x; i < H_DIM; i += 256) {
    int key = (int)(p[i] * 2048.0f);
    key = key < 0 ? 0 : (key > 4095 ? 4095 : key);
    atomicAdd(&hist[key], 1);
  }
  __syncthreads();

  if (threadIdx.x == 0) {
    int cum = 0, thr = 0, above = 0;
    for (int b = 4095; b >= 0; --b) {
      int c = hist[b];
      if (cum + c >= TOPK) { thr = b; above = cum; break; }
      cum += c;
    }
    s_thr = thr; s_above = above;
  }
  __syncthreads();
  int thr = s_thr, above = s_above;

  for (int i = threadIdx.x; i < H_DIM; i += 256) {
    float v = p[i];
    int key = (int)(v * 2048.0f);
    key = key < 0 ? 0 : (key > 4095 ? 4095 : key);
    bool keep = false;
    if (key > thr) keep = true;
    else if (key == thr) {
      int e = atomicAdd(&s_eq, 1);
      if (e < TOPK - above) keep = true;
    }
    if (keep) {
      int slot = atomicAdd(&s_slot, 1);
      idx_list[(size_t)row * TOPK + slot] = i;
      val_list[(size_t)row * TOPK + slot] = v;
    } else {
      p[i] = 0.0f;
    }
  }
}

// ---------------------------------------------------------------------------
// 5. Tiled transpose W [D,H] -> WT [H,D] (coalesced decoder gathers, L2-hot)
// ---------------------------------------------------------------------------
__global__ __launch_bounds__(256) void transpose_w(const float* __restrict__ vd_w,
                                                   const float* __restrict__ td_w,
                                                   float* __restrict__ WT) {
  int m = blockIdx.z;
  const float* W = m ? td_w : vd_w;
  float* O = WT + (size_t)m * H_DIM * D_DIM;
  int h0 = blockIdx.x * 32, d0 = blockIdx.y * 32;
  __shared__ float tile[32][33];
  int tid = threadIdx.x;
  #pragma unroll
  for (int p = 0; p < 4; ++p) {
    int di = p * 8 + (tid >> 5), hj = tid & 31;
    tile[di][hj] = W[(size_t)(d0 + di) * H_DIM + h0 + hj];
  }
  __syncthreads();
  #pragma unroll
  for (int p = 0; p < 4; ++p) {
    int hi = p * 8 + (tid >> 5), dj = tid & 31;
    O[(size_t)(h0 + hi) * D_DIM + d0 + dj] = tile[dj][hi];
  }
}

// ---------------------------------------------------------------------------
// 6. Sparse decode: recon[b,:] = bias + sum_k val_k * WT[idx_k, :]
// ---------------------------------------------------------------------------
__global__ __launch_bounds__(256) void decode(const float* __restrict__ WT,
                                              const int* __restrict__ idx_list,
                                              const float* __restrict__ val_list,
                                              const float* __restrict__ vd_b,
                                              const float* __restrict__ td_b,
                                              float* __restrict__ recon) {
  int b = blockIdx.x, m = blockIdx.y;
  const float* bias = m ? td_b : vd_b;
  const float* wt = WT + (size_t)m * H_DIM * D_DIM;
  int row = m * B_DIM + b;
  __shared__ int   sidx[TOPK];
  __shared__ float sval[TOPK];
  if (threadIdx.x < TOPK) {
    sidx[threadIdx.x] = idx_list[(size_t)row * TOPK + threadIdx.x];
    sval[threadIdx.x] = val_list[(size_t)row * TOPK + threadIdx.x];
  }
  __syncthreads();
  float* out = recon + ((size_t)m * B_DIM + b) * D_DIM;
  for (int d = threadIdx.x; d < D_DIM; d += 256) {
    float acc = bias[d];
    #pragma unroll 8
    for (int k = 0; k < TOPK; ++k)
      acc += sval[k] * wt[(size_t)sidx[k] * D_DIM + d];
    out[d] = acc;
  }
}

// ---------------------------------------------------------------------------
extern "C" void kernel_launch(void* const* d_in, const int* in_sizes, int n_in,
                              void* d_out, int out_size, void* d_ws, size_t ws_size,
                              hipStream_t stream) {
  const float* v    = (const float*)d_in[0];
  const float* t    = (const float*)d_in[1];
  const float* enc  = (const float*)d_in[2];
  const float* vd_w = (const float*)d_in[3];
  const float* vd_b = (const float*)d_in[4];
  const float* td_w = (const float*)d_in[5];
  const float* td_b = (const float*)d_in[6];
  float* out = (float*)d_out;

  char* ws = (char*)d_ws;
  size_t off = 0;
  _Float16* ef16 = (_Float16*)(ws + off); off += (size_t)2 * B_DIM * D_DIM * 2;  // 16.8 MB
  _Float16* wTs  = (_Float16*)(ws + off); off += (size_t)D_DIM * H_DIM * 2;      // 33.6 MB
  float*    WT   = (float*)   (ws + off); off += (size_t)2 * H_DIM * D_DIM * 4;  // 134 MB
  int*   idx_list = (int*)  (ws + off); off += (size_t)2 * B_DIM * TOPK * 4;
  float* val_list = (float*)(ws + off); off += (size_t)2 * B_DIM * TOPK * 4;

  float* recon  = out;                                // [2][B,D]
  float* latent = out + (size_t)2 * B_DIM * D_DIM;    // [2][B,H]

  prep_emb    <<<dim3(B_DIM, 2),            256, 0, stream>>>(v, t, ef16);
  prep_encoder<<<dim3(H_DIM / 32),          256, 0, stream>>>(enc, wTs);
  transpose_w <<<dim3(H_DIM / 32, D_DIM / 32, 2), 256, 0, stream>>>(vd_w, td_w, WT);
  gemm_act    <<<dim3(H_DIM / 256, B_DIM / 128, 2), 256, 0, stream>>>(ef16, wTs, latent);
  topk_select <<<dim3(2 * B_DIM),           256, 0, stream>>>(latent, idx_list, val_list);
  decode      <<<dim3(B_DIM, 2),            256, 0, stream>>>(WT, idx_list, val_list, vd_b, td_b, recon);
}
